// NonLocalSelfAttention_58076547777045
// MI455X (gfx1250) — matrix operand order repaired and checked by reading, not compiled, not run on hardware
//
#include <hip/hip_runtime.h>
#include <math.h>

typedef __attribute__((ext_vector_type(16))) _Float16 v16h;
typedef __attribute__((ext_vector_type(8)))  float    v8f;

#define N_TOK   8192      // n*h*w
#define DH      128       // Ch
#define C_IN    256
#define SPAT    4096      // h*w
#define BN      64        // key chunk per iteration
#define LDK     (DH + 8)  // LDS row stride (halfs) for K chunk
#define LDV     (BN + 8)  // LDS row stride for V chunk (transposed)
#define LDP     (BN + 8)  // LDS row stride for per-wave P tile

// ---------------------------------------------------------------------------
// Kernel A: 1x1-conv projections X(8192x256) -> q,k,v (8192x128), f32 -> f16.
//   attn uses: scores = k @ q^T, so   attn-Q := k_proj, attn-K := q_proj.
//   Ka : attn keys   (q_proj) row-major [N][DH]
//   Qa : attn query  (k_proj) row-major [N][DH]
//   Vt : values transposed [DH][N]  (so flash-V B-frags read contiguously)
// ---------------------------------------------------------------------------
__global__ __launch_bounds__(128)
void qkv_kernel(const float* __restrict__ feat,
                const float* __restrict__ Wq, const float* __restrict__ bq,
                const float* __restrict__ Wk, const float* __restrict__ bk,
                const float* __restrict__ Wv, const float* __restrict__ bv,
                _Float16* __restrict__ Ka,
                _Float16* __restrict__ Qa,
                _Float16* __restrict__ Vt)
{
    const int lane = threadIdx.x & 31;
    const int wv   = threadIdx.x >> 5;
    const int lm   = lane & 15;
    const int hi   = lane >> 4;
    const int row0 = (blockIdx.x * 4 + wv) * 16;

    // A fragments: 16 rows of X over all 256 channels (8 k-steps of 32).
    // X[p][c] = feat[img*C*SPAT + c*SPAT + sp],  p = img*SPAT + sp
    v16h af[8];
    {
        const int m   = row0 + lm;          // A layout: M = lane%16 (both halves)
        const int img = m >> 12;
        const int sp  = m & (SPAT - 1);
        const size_t base = (size_t)img * C_IN * SPAT + sp;
        #pragma unroll
        for (int kk = 0; kk < 8; ++kk) {
            #pragma unroll
            for (int v = 0; v < 8; ++v) {
                const int kb = kk * 32 + hi * 16 + v * 2;
                af[kk][2*v]   = (_Float16)feat[base + (size_t)kb * SPAT];
                af[kk][2*v+1] = (_Float16)feat[base + (size_t)(kb + 1) * SPAT];
            }
        }
    }

    const float* Wmat[3] = {Wq, Wk, Wv};
    const float* Bvec[3] = {bq, bk, bv};
    const v8f zero = {};

    #pragma unroll
    for (int mat = 0; mat < 3; ++mat) {
        const float* W = Wmat[mat];         // [128][256] row-major
        v8f acc[8];
        #pragma unroll
        for (int t = 0; t < 8; ++t) acc[t] = zero;

        #pragma unroll
        for (int kk = 0; kk < 8; ++kk) {
            #pragma unroll
            for (int t = 0; t < 8; ++t) {
                // B layout: N = lane%16 (output channel), K pairs at hi*16+2v
                const int o = t * 16 + lm;
                const float* wrow = W + (size_t)o * C_IN + kk * 32 + hi * 16;
                v16h bf;
                #pragma unroll
                for (int v = 0; v < 8; ++v) {
                    bf[2*v]   = (_Float16)wrow[2*v];
                    bf[2*v+1] = (_Float16)wrow[2*v+1];
                }
                acc[t] = __builtin_amdgcn_wmma_f32_16x16x32_f16(
                    false, af[kk], false, bf, (short)0, acc[t], false, false);
            }
        }

        // C layout: row = row0 + hi*8 + r, col = t*16 + lane%16
        const int mbase = row0 + hi * 8;
        #pragma unroll
        for (int t = 0; t < 8; ++t) {
            const int o = t * 16 + lm;
            const float bias = Bvec[mat][o];
            #pragma unroll
            for (int r = 0; r < 8; ++r) {
                const _Float16 hval = (_Float16)(acc[t][r] + bias);
                const int p = mbase + r;
                if (mat == 0)      Ka[(size_t)p * DH + o]   = hval; // q_proj
                else if (mat == 1) Qa[(size_t)p * DH + o]   = hval; // k_proj
                else               Vt[(size_t)o * N_TOK + p] = hval; // v^T
            }
        }
    }
}

// ---------------------------------------------------------------------------
// Kernel B: flash attention.  softmax(Qa @ Ka^T) @ V, d = 128, no scale
// (reference applies no 1/sqrt(d)).  One wave = 16 query rows; WG = 64 rows.
// K/V chunks are staged memory->LDS with GLOBAL_LOAD_ASYNC_TO_LDS_B128
// (ASYNCcnt path, no VGPR round trip); next chunk is prefetched into cache
// during compute so the async loads of the following iteration hit L0/L2.
// ---------------------------------------------------------------------------
__global__ __launch_bounds__(128)
void flash_kernel(const _Float16* __restrict__ Qa,
                  const _Float16* __restrict__ Ka,
                  const _Float16* __restrict__ Vt,
                  float* __restrict__ Mout)
{
    __shared__ _Float16 Ks[BN][LDK];      // K chunk, row-major [j][d]
    __shared__ _Float16 Vs[DH][LDV];      // V chunk, transposed [d][j]
    __shared__ _Float16 Pb[4][16][LDP];   // per-wave P tile [m][j]

    const int tid  = threadIdx.x;
    const int lane = tid & 31;
    const int wv   = tid >> 5;
    const int lm   = lane & 15;
    const int hi   = lane >> 4;
    const int row0 = (blockIdx.x * 4 + wv) * 16;

    // Per-thread staging addresses (constant across the j0 loop except j0):
    const int krow = tid >> 1;            // K chunk row this thread stages
    const int koff = (tid & 1) * 64;      // which 64-half of the row
    const unsigned ks_lds = (unsigned)(size_t)(&Ks[krow][koff]);
    const unsigned vs_lds = (unsigned)(size_t)(&Vs[tid][0]);

    // Preload Q A-fragments (16 rows x 128 d = 4 k-steps), resident all loop.
    v16h qf[4];
    #pragma unroll
    for (int kk = 0; kk < 4; ++kk) {
        const _Float16* qr = Qa + (size_t)(row0 + lm) * DH;
        #pragma unroll
        for (int v = 0; v < 8; ++v) {
            const int kb = kk * 32 + hi * 16 + v * 2;
            qf[kk][2*v]   = qr[kb];
            qf[kk][2*v+1] = qr[kb + 1];
        }
    }

    const v8f zero = {};
    v8f O[8];                               // 16 x 128 f32 accumulator
    #pragma unroll
    for (int t = 0; t < 8; ++t) O[t] = zero;
    float mprev[8], lsum[8];
    #pragma unroll
    for (int r = 0; r < 8; ++r) { mprev[r] = -1.0e30f; lsum[r] = 0.0f; }

    for (int j0 = 0; j0 < N_TOK; j0 += BN) {
        // ---- async stage of K chunk (row-major) and V chunk (transposed) ----
        {
            const _Float16* gk = Ka + (size_t)(j0 + krow) * DH + koff;
            #pragma unroll
            for (int i = 0; i < 8; ++i) {
                asm volatile("global_load_async_to_lds_b128 %0, %1, off"
                             :: "v"(ks_lds + (unsigned)(i * 16)),
                                "v"(gk + i * 8)
                             : "memory");
            }
            const _Float16* gv = Vt + (size_t)tid * N_TOK + j0;
            #pragma unroll
            for (int i = 0; i < 8; ++i) {
                asm volatile("global_load_async_to_lds_b128 %0, %1, off"
                             :: "v"(vs_lds + (unsigned)(i * 16)),
                                "v"(gv + i * 8)
                             : "memory");
            }
        }
        asm volatile("s_wait_asynccnt 0x0" ::: "memory");
        __syncthreads();

        // Prefetch next chunk into cache while we compute on this one.
        if (j0 + BN < N_TOK) {
            __builtin_prefetch(Ka + (size_t)(j0 + BN + krow) * DH + koff, 0, 3);
            __builtin_prefetch(Vt + (size_t)tid * N_TOK + (j0 + BN), 0, 3);
        }

        // S (16 x 64) = Q_tile (16 x 128) @ Kchunk^T (128 x 64)
        v8f S[4];
        #pragma unroll
        for (int t = 0; t < 4; ++t) S[t] = zero;
        #pragma unroll
        for (int kk = 0; kk < 4; ++kk) {
            #pragma unroll
            for (int t = 0; t < 4; ++t) {
                const _Float16* kr = &Ks[t * 16 + lm][kk * 32 + hi * 16];
                v16h bf;
                #pragma unroll
                for (int v = 0; v < 8; ++v) {
                    bf[2*v]   = kr[2*v];
                    bf[2*v+1] = kr[2*v+1];
                }
                S[t] = __builtin_amdgcn_wmma_f32_16x16x32_f16(
                    false, qf[kk], false, bf, (short)0, S[t], false, false);
            }
        }

        // Online softmax: row r of the tile lives in one 16-lane half.
        float cmax[8];
        #pragma unroll
        for (int r = 0; r < 8; ++r)
            cmax[r] = fmaxf(fmaxf(S[0][r], S[1][r]), fmaxf(S[2][r], S[3][r]));
        #pragma unroll
        for (int off = 1; off < 16; off <<= 1) {
            #pragma unroll
            for (int r = 0; r < 8; ++r)
                cmax[r] = fmaxf(cmax[r], __shfl_xor(cmax[r], off, 32));
        }
        float scale[8], rsum[8];
        #pragma unroll
        for (int r = 0; r < 8; ++r) {
            const float mn = fmaxf(mprev[r], cmax[r]);
            scale[r] = __expf(mprev[r] - mn);
            mprev[r] = mn;
            rsum[r]  = 0.0f;
        }
        // exp + park P in per-wave LDS tile (C-layout -> memory [m][j])
        #pragma unroll
        for (int t = 0; t < 4; ++t) {
            #pragma unroll
            for (int r = 0; r < 8; ++r) {
                const float p = __expf(S[t][r] - mprev[r]);
                rsum[r] += p;
                Pb[wv][hi * 8 + r][t * 16 + lm] = (_Float16)p;
            }
        }
        #pragma unroll
        for (int off = 1; off < 16; off <<= 1) {
            #pragma unroll
            for (int r = 0; r < 8; ++r)
                rsum[r] += __shfl_xor(rsum[r], off, 32);
        }
        #pragma unroll
        for (int r = 0; r < 8; ++r) lsum[r] = lsum[r] * scale[r] + rsum[r];
        #pragma unroll
        for (int t = 0; t < 8; ++t) {
            #pragma unroll
            for (int r = 0; r < 8; ++r) O[t][r] *= scale[r];
        }

        __builtin_amdgcn_wave_barrier();   // order P stores before A-frag reloads

        // O (16 x 128) += P (16 x 64) @ Vchunk (64 x 128)
        #pragma unroll
        for (int kk = 0; kk < 2; ++kk) {
            v16h pf;                        // A-frag of P from LDS
            #pragma unroll
            for (int v = 0; v < 8; ++v) {
                const int jb = kk * 32 + hi * 16 + v * 2;
                pf[2*v]   = Pb[wv][lm][jb];
                pf[2*v+1] = Pb[wv][lm][jb + 1];
            }
            #pragma unroll
            for (int t = 0; t < 8; ++t) {
                const _Float16* vr = &Vs[t * 16 + lm][kk * 32 + hi * 16];
                v16h bf;                    // B-frag of V: contiguous in j
                #pragma unroll
                for (int v = 0; v < 8; ++v) {
                    bf[2*v]   = vr[2*v];
                    bf[2*v+1] = vr[2*v+1];
                }
                O[t] = __builtin_amdgcn_wmma_f32_16x16x32_f16(
                    false, pf, false, bf, (short)0, O[t], false, false);
            }
        }
        __syncthreads();
    }

    // Normalize and write mask rows (f32, row-major [N][DH]).
    float inv[8];
    #pragma unroll
    for (int r = 0; r < 8; ++r) inv[r] = 1.0f / lsum[r];
    #pragma unroll
    for (int t = 0; t < 8; ++t) {
        #pragma unroll
        for (int r = 0; r < 8; ++r) {
            const int p = row0 + hi * 8 + r;
            const int d = t * 16 + lm;
            Mout[(size_t)p * DH + d] = O[t][r] * inv[r];
        }
    }
}

// ---------------------------------------------------------------------------
// Kernel C: out[n,co,a,b] = Wm[co,:] . M[n*4096 + b*64 + a, :] + bm[co]
//                           + feature[n,co,a,b]
// (the reference's (0,3,2,1) transpose chain folds into the p index).
// ---------------------------------------------------------------------------
__global__ __launch_bounds__(256)
void outproj_kernel(const float* __restrict__ Mbuf,
                    const float* __restrict__ Wm,
                    const float* __restrict__ bm,
                    const float* __restrict__ feat,
                    float* __restrict__ out)
{
    __shared__ float mrow[DH];
    const int img = blockIdx.x >> 6;
    const int a   = blockIdx.x & 63;
    const int co  = threadIdx.x;

    float wreg[DH];
    #pragma unroll
    for (int ch = 0; ch < DH; ++ch) wreg[ch] = Wm[(size_t)co * DH + ch];
    const float bias = bm[co];

    for (int b = 0; b < 64; ++b) {
        const int p = img * SPAT + b * 64 + a;
        if (threadIdx.x < DH) mrow[threadIdx.x] = Mbuf[(size_t)p * DH + threadIdx.x];
        __syncthreads();
        float s = bias;
        #pragma unroll
        for (int ch = 0; ch < DH; ++ch) s = fmaf(wreg[ch], mrow[ch], s);
        const size_t oidx = (((size_t)img * C_IN + co) * 64 + a) * 64 + b;
        out[oidx] = s + feat[oidx];
        __syncthreads();
    }
}

// ---------------------------------------------------------------------------
extern "C" void kernel_launch(void* const* d_in, const int* in_sizes, int n_in,
                              void* d_out, int out_size, void* d_ws, size_t ws_size,
                              hipStream_t stream)
{
    const float* feat = (const float*)d_in[0];
    const float* Wq   = (const float*)d_in[1];
    const float* bq   = (const float*)d_in[2];
    const float* Wk   = (const float*)d_in[3];
    const float* bk   = (const float*)d_in[4];
    const float* Wv   = (const float*)d_in[5];
    const float* bv   = (const float*)d_in[6];
    const float* Wm   = (const float*)d_in[7];
    const float* bm   = (const float*)d_in[8];
    float* out = (float*)d_out;

    // Workspace layout: Ka | Qa | Vt (f16) | Mbuf (f32)  => ~10.5 MB total
    _Float16* Ka = (_Float16*)d_ws;
    _Float16* Qa = Ka + (size_t)N_TOK * DH;
    _Float16* Vt = Qa + (size_t)N_TOK * DH;
    float*    Mb = (float*)(Vt + (size_t)DH * N_TOK);

    hipLaunchKernelGGL(qkv_kernel, dim3(N_TOK / 64), dim3(128), 0, stream,
                       feat, Wq, bq, Wk, bk, Wv, bv, Ka, Qa, Vt);
    hipLaunchKernelGGL(flash_kernel, dim3(N_TOK / 64), dim3(128), 0, stream,
                       Qa, Ka, Vt, Mb);
    hipLaunchKernelGGL(outproj_kernel, dim3(2 * 64), dim3(256), 0, stream,
                       Mb, Wm, bm, feat, out);
}
